// DayAdapter_87058987089974
// MI455X (gfx1250) — matrix operand order repaired and checked by reading, not compiled, not run on hardware
//
#include <hip/hip_runtime.h>
#include <hip/hip_bf16.h>

typedef __attribute__((ext_vector_type(16))) _Float16 v16h;
typedef __attribute__((ext_vector_type(8)))  _Float16 v8h;
typedef __attribute__((ext_vector_type(8)))  float    v8f;

#define B_    64
#define T_    512
#define DIN   512
#define DHID  1024
#define DOUT  512
#define HS_STRIDE 1032   // 1024 + 8 halves pad -> 2064B row stride, bank stride 4, 16B aligned

#if __has_builtin(__builtin_amdgcn_sched_group_barrier)
#define SGB(mask, cnt) __builtin_amdgcn_sched_group_barrier(mask, cnt, 0)
#else
#define SGB(mask, cnt)
#endif
// mask bits: 0x008 = MFMA/WMMA, 0x020 = VMEM read, 0x100 = DS read

// ---------------------------------------------------------------------------
// x: f32 -> f16, 8 elements per thread, coalesced
// ---------------------------------------------------------------------------
__global__ __launch_bounds__(256) void cvt_x_kernel(const float* __restrict__ x,
                                                    _Float16* __restrict__ xh, long n8) {
    long i = (long)blockIdx.x * blockDim.x + threadIdx.x;
    if (i >= n8) return;
    const float4* p = (const float4*)(x + i * 8);
    float4 a = p[0], b = p[1];
    v8h o;
    o[0] = (_Float16)a.x; o[1] = (_Float16)a.y; o[2] = (_Float16)a.z; o[3] = (_Float16)a.w;
    o[4] = (_Float16)b.x; o[5] = (_Float16)b.y; o[6] = (_Float16)b.z; o[7] = (_Float16)b.w;
    *(v8h*)(xh + i * 8) = o;
}

// ---------------------------------------------------------------------------
// Pack weights [days, K, N] f32 row-major into WMMA B-fragment order, f16.
// Packet index p = ((day*kkTiles + kk)*niTiles + ni)*32 + lane, 16 halves/packet.
// B-frag (32x16 f16): lane holds col N = lane&15; halves i<8: K = i + 8*(lane>>4), i>=8: K = 16 + (i-8) + 8*(lane>>4).
// ---------------------------------------------------------------------------
__global__ __launch_bounds__(256) void pack_w_kernel(const float* __restrict__ W,
                                                     _Float16* __restrict__ Wp,
                                                     int K, int N, long npkt) {
    long p = (long)blockIdx.x * blockDim.x + threadIdx.x;
    if (p >= npkt) return;
    int lane = (int)(p & 31);
    long t = p >> 5;
    int niTiles = N >> 4, kkTiles = K >> 5;
    int ni  = (int)(t % niTiles); t /= niTiles;
    int kk  = (int)(t % kkTiles);
    int day = (int)(t / kkTiles);
    int n     = ni * 16 + (lane & 15);
    int kbase = kk * 32 + ((lane >> 4) << 3);
    const float* src = W + ((long)day * K + kbase) * N + n;
    v8h lo, hi;
#pragma unroll
    for (int i = 0; i < 8; ++i) lo[i] = (_Float16)src[(long)i * N];
#pragma unroll
    for (int i = 0; i < 8; ++i) hi[i] = (_Float16)src[(long)(i + 16) * N];
    v8h* dst = (v8h*)(Wp + p * 16);
    dst[0] = lo;
    dst[1] = hi;
}

__device__ __forceinline__ v16h cat16(v8h lo, v8h hi) {
    return __builtin_shufflevector(lo, hi, 0,1,2,3,4,5,6,7,8,9,10,11,12,13,14,15);
}

// ---------------------------------------------------------------------------
// Fused adapter: one workgroup = 16 tokens of one sample. 8 waves.
// Stage1: H(16x1024) = relu(X(16x512) @ W1 + b1) -> LDS f16
// Stage2: Y(16x512)  = H @ W2 + b2 -> global f32 (pre-LayerNorm)
// sched_group_barrier forces per-K-step shape: [all frag loads] -> [WMMA burst]
// so only one load-latency exposure per K-step and 8 live B fragments.
// ---------------------------------------------------------------------------
__global__ __launch_bounds__(256) void adapter_main_kernel(
    const _Float16* __restrict__ xh, const int* __restrict__ day_idx,
    const _Float16* __restrict__ w1p, const float* __restrict__ b1,
    const _Float16* __restrict__ w2p, const float* __restrict__ b2,
    float* __restrict__ out) {
    __shared__ __align__(16) _Float16 Hs[16 * HS_STRIDE + 64];  // +64: prefetch overread pad

    const int b    = blockIdx.y;
    const int t0   = blockIdx.x * 16;
    const int day  = day_idx[b];
    const int tid  = threadIdx.x;
    const int wave = tid >> 5;
    const int lane = tid & 31;
    const int nsel   = lane & 15;          // N col within tile / A row M
    const int rowoff = (lane >> 4) << 3;   // C/D: rows 0..7 or 8..15
    const int kband  = (lane >> 4) << 3;   // A frag K-base offset (0 or 8)

    const _Float16* xrow = xh + ((long)(b * T_) + t0) * DIN;
    const _Float16* w1d  = w1p + (long)day * (DIN  / 32) * (DHID / 16) * 512;
    const _Float16* w2d  = w2p + (long)day * (DHID / 32) * (DOUT / 16) * 512;

    // ---------------- Stage 1 ----------------
    {
        const int ni0 = wave * 8;          // 8 N-tiles per wave (8*8*16 = 1024 cols)
        const _Float16* abase = xrow + (long)nsel * DIN + kband;        // + kk*32
        const _Float16* bbase = w1d + ((long)ni0) * 512 + lane * 16;    // + kk*(DHID/16)*512 + j*512

        v8f acc[8];
#pragma unroll
        for (int j = 0; j < 8; ++j) {
            float bv = b1[day * DHID + (ni0 + j) * 16 + nsel];
#pragma unroll
            for (int r = 0; r < 8; ++r) acc[j][r] = bv;
        }

        // prologue: A fragment for kk = 0
        v8h alo = *(const v8h*)(abase);
        v8h ahi = *(const v8h*)(abase + 16);
        for (int kk = 0; kk < DIN / 32; ++kk) {
            const _Float16* bp = bbase + (long)kk * (DHID / 16) * 512;
            v16h bf[8];
#pragma unroll
            for (int j = 0; j < 8; ++j) {
                v8h blo = *(const v8h*)(bp + j * 512);
                v8h bhi = *(const v8h*)(bp + j * 512 + 8);
                bf[j] = cat16(blo, bhi);
            }
            v16h a = cat16(alo, ahi);
            // unconditional next-A prefetch (last iter overreads into ws: harmless)
            {
                const _Float16* ap = abase + (kk + 1) * 32;
                alo = *(const v8h*)(ap);
                ahi = *(const v8h*)(ap + 16);
            }
#pragma unroll
            for (int j = 0; j < 8; ++j) {
                acc[j] = __builtin_amdgcn_wmma_f32_16x16x32_f16(
                    false, a, false, bf[j], (short)0, acc[j], false, false);
            }
            SGB(0x020, 18);   // 16 B-frag + 2 A-prefetch vmem reads first
            SGB(0x008, 8);    // then the 8-WMMA burst
        }
#pragma unroll
        for (int j = 0; j < 8; ++j) {
            int n = (ni0 + j) * 16 + nsel;
#pragma unroll
            for (int r = 0; r < 8; ++r) {
                float v = acc[j][r];
                v = v > 0.0f ? v : 0.0f;
                Hs[(r + rowoff) * HS_STRIDE + n] = (_Float16)v;
            }
        }
    }
    __syncthreads();
    // ---------------- Stage 2 ----------------
    {
        const int ni0 = wave * 4;          // 4 N-tiles per wave (8*4*16 = 512 cols)
        const _Float16* abase = &Hs[nsel * HS_STRIDE + kband];          // + kk*32 (LDS)
        const _Float16* bbase = w2d + ((long)ni0) * 512 + lane * 16;

        v8f acc[4];
#pragma unroll
        for (int j = 0; j < 4; ++j) {
            float bv = b2[day * DOUT + (ni0 + j) * 16 + nsel];
#pragma unroll
            for (int r = 0; r < 8; ++r) acc[j][r] = bv;
        }

        v8h alo = *(const v8h*)(abase);
        v8h ahi = *(const v8h*)(abase + 16);
        for (int kk = 0; kk < DHID / 32; ++kk) {
            const _Float16* bp = bbase + (long)kk * (DOUT / 16) * 512;
            v16h bf[4];
#pragma unroll
            for (int j = 0; j < 4; ++j) {
                v8h blo = *(const v8h*)(bp + j * 512);
                v8h bhi = *(const v8h*)(bp + j * 512 + 8);
                bf[j] = cat16(blo, bhi);
            }
            v16h a = cat16(alo, ahi);
            // unconditional next-A prefetch from LDS (last iter stays in padded Hs)
            {
                const _Float16* ap = abase + (kk + 1) * 32;
                alo = *(const v8h*)(ap);
                ahi = *(const v8h*)(ap + 16);
            }
#pragma unroll
            for (int j = 0; j < 4; ++j) {
                acc[j] = __builtin_amdgcn_wmma_f32_16x16x32_f16(
                    false, a, false, bf[j], (short)0, acc[j], false, false);
            }
            SGB(0x020, 8);    // 8 B-frag vmem reads
            SGB(0x100, 2);    // 2 A-prefetch DS reads
            SGB(0x008, 4);    // 4-WMMA burst
        }
#pragma unroll
        for (int j = 0; j < 4; ++j) {
            int n = (ni0 + j) * 16 + nsel;
#pragma unroll
            for (int r = 0; r < 8; ++r) {
                out[((long)(b * T_) + t0 + r + rowoff) * DOUT + n] = acc[j][r];
            }
        }
    }
}

// ---------------------------------------------------------------------------
// In-place LayerNorm over last dim (512) + per-day gamma/beta.
// One wave32 per row; float4 I/O; shfl_xor wave reduction.
// ---------------------------------------------------------------------------
__global__ __launch_bounds__(256) void ln_kernel(float* __restrict__ y,
                                                 const int* __restrict__ day_idx,
                                                 const float* __restrict__ gamma,
                                                 const float* __restrict__ beta) {
    const int row  = blockIdx.x * 8 + (threadIdx.x >> 5);
    const int lane = threadIdx.x & 31;
    const int b    = row >> 9;             // row / T_
    const int day  = day_idx[b];
    float* yr = y + (long)row * DOUT;

    float4 v[4];
    float s = 0.0f, ss = 0.0f;
#pragma unroll
    for (int q = 0; q < 4; ++q) {
        v[q] = *(const float4*)(yr + (q * 32 + lane) * 4);
        s  += v[q].x + v[q].y + v[q].z + v[q].w;
        ss += v[q].x * v[q].x + v[q].y * v[q].y + v[q].z * v[q].z + v[q].w * v[q].w;
    }
#pragma unroll
    for (int off = 16; off > 0; off >>= 1) {
        s  += __shfl_xor(s,  off, 32);
        ss += __shfl_xor(ss, off, 32);
    }
    const float mean = s * (1.0f / DOUT);
    const float var  = ss * (1.0f / DOUT) - mean * mean;
    const float rstd = rsqrtf(var + 1e-5f);
    const float* g  = gamma + (long)day * DOUT;
    const float* be = beta  + (long)day * DOUT;
#pragma unroll
    for (int q = 0; q < 4; ++q) {
        int c = (q * 32 + lane) * 4;
        float4 gv = *(const float4*)(g + c);
        float4 bv = *(const float4*)(be + c);
        float4 o;
        o.x = (v[q].x - mean) * rstd * gv.x + bv.x;
        o.y = (v[q].y - mean) * rstd * gv.y + bv.y;
        o.z = (v[q].z - mean) * rstd * gv.z + bv.z;
        o.w = (v[q].w - mean) * rstd * gv.w + bv.w;
        *(float4*)(yr + c) = o;
    }
}

// ---------------------------------------------------------------------------
extern "C" void kernel_launch(void* const* d_in, const int* in_sizes, int n_in,
                              void* d_out, int out_size, void* d_ws, size_t ws_size,
                              hipStream_t stream) {
    const float* x     = (const float*)d_in[0];
    const int*   day   = (const int*)  d_in[1];
    const float* W1    = (const float*)d_in[2];
    const float* b1    = (const float*)d_in[3];
    const float* W2    = (const float*)d_in[4];
    const float* b2    = (const float*)d_in[5];
    const float* gamma = (const float*)d_in[6];
    const float* beta  = (const float*)d_in[7];
    float* out = (float*)d_out;

    const long xn  = (long)B_ * T_ * DIN;          // 16,777,216 elems
    const long w1n = (long)24 * DIN * DHID;        // 12,582,912 elems
    const long w2n = (long)24 * DHID * DOUT;       // 12,582,912 elems
    const size_t xh_bytes  = (size_t)xn  * 2;      // 33.5 MB
    const size_t w1p_bytes = (size_t)w1n * 2;      // 25.2 MB
    const size_t w2p_bytes = (size_t)w2n * 2;      // 25.2 MB
    if (ws_size < xh_bytes + w1p_bytes + w2p_bytes) return;  // ws too small: fail loudly

    char* ws = (char*)d_ws;
    _Float16* xh  = (_Float16*)ws;
    _Float16* w1p = (_Float16*)(ws + xh_bytes);
    _Float16* w2p = (_Float16*)(ws + xh_bytes + w1p_bytes);

    {   // x conversion: 8 elems/thread
        long n8 = xn / 8;
        cvt_x_kernel<<<(unsigned)((n8 + 255) / 256), 256, 0, stream>>>(x, xh, n8);
    }
    {   // weight packing: 16 halves/thread
        long p1 = w1n / 16, p2 = w2n / 16;
        pack_w_kernel<<<(unsigned)((p1 + 255) / 256), 256, 0, stream>>>(W1, w1p, DIN,  DHID, p1);
        pack_w_kernel<<<(unsigned)((p2 + 255) / 256), 256, 0, stream>>>(W2, w2p, DHID, DOUT, p2);
    }
    {   // fused adapter GEMMs
        dim3 grid(T_ / 16, B_);
        adapter_main_kernel<<<grid, 256, 0, stream>>>(xh, day, w1p, b1, w2p, b2, out);
    }
    {   // layernorm, one wave per row
        ln_kernel<<<(B_ * T_) / 8, 256, 0, stream>>>(out, day, gamma, beta);
    }
}
